// RIM_GRU_73821897884330
// MI455X (gfx1250) — matrix operand order repaired and checked by reading, not compiled, not run on hardware
//
#include <hip/hip_runtime.h>
#include <cstddef>

typedef __bf16 bf16;
typedef __attribute__((ext_vector_type(16))) __bf16 v16bf;
typedef __attribute__((ext_vector_type(8)))  __bf16 v8bf;
typedef __attribute__((ext_vector_type(8)))  float  v8f;

// ---------------- problem constants ----------------
constexpr int kT = 256, kB = 128, kNINP = 512;
constexpr int kNB = 6, kBS = 100, kTOPK = 4, kH = 600;
constexpr int kKD = 64, kVD = 100, kCH = 4, kCK = 32;
// padded (WMMA-friendly) dims
constexpr int kBSP = 128;   // BS padded to K-multiple-of-32
constexpr int kG3P = 304;   // 3*BS padded to N-multiple-of-16
constexpr int kVP  = 112;   // VD padded (N of Wv)
constexpr int kCKV = 128;   // CH*CK == CH*CV
constexpr int kCOP = 112;   // Wco output padded
constexpr int kMB  = 16;    // batch rows per workgroup
constexpr int kThreads = 256;  // 8 waves (wave32)
constexpr int kTile = 512;  // elements per swizzled 32x16 B tile

// ---------------- bf16 workspace layout (element offsets) ----------------
// All matrices stored pre-swizzled in WMMA B-fragment order:
//   tile(nt,kt) at ((nt*KT)+kt)*512, lane l owns 16 contiguous bf16 =
//   column n = nt*16+(l&15), K = kt*32+(l>>4)*8+{0..7,16..23}.
constexpr size_t OFF_WK  = 0;
constexpr size_t OFF_WV  = OFF_WK  + (size_t)512 * 64;
constexpr size_t OFF_WQ  = OFF_WV  + (size_t)512 * kVP;
constexpr size_t OFF_WIH = OFF_WQ  + (size_t)kBSP * 64;
constexpr size_t OFF_WHH = OFF_WIH + (size_t)kNB * kBSP * kG3P;
constexpr size_t OFF_WCQ = OFF_WHH + (size_t)kNB * kBSP * kG3P;
constexpr size_t OFF_WCK = OFF_WCQ + (size_t)kBSP * kCKV;
constexpr size_t OFF_WCV = OFF_WCK + (size_t)kBSP * kCKV;
constexpr size_t OFF_WCO = OFF_WCV + (size_t)kBSP * kCKV;

// ---------------- WMMA fragment helpers (wave32 layouts per CDNA5 ISA) ----
union FragU { v16bf v; v8bf h[2]; };

// A: 16x32 bf16 from LDS. Lane<16: row M=lane, K {0..7,16..23}; lane>=16:
// row M=lane-16, K {8..15,24..31}. Two contiguous 16-byte runs -> 2x b128.
__device__ __forceinline__ v16bf load_A(const bf16* A, int lda, int kbase, int lane) {
  const int row = lane & 15;
  const int kh  = (lane >> 4) & 1;
  const bf16* p = A + row * lda + kbase + kh * 8;
  FragU u;
  u.h[0] = *(const v8bf*)(p);
  u.h[1] = *(const v8bf*)(p + 16);
  return u.v;
}
__device__ __forceinline__ v8f wmma_bf16(v16bf a, v16bf b, v8f c) {
  return __builtin_amdgcn_wmma_f32_16x16x32_bf16(false, a, false, b, (short)0, c,
                                                 false, false);
}
// Bnt points at the first k-tile of this n-tile (swizzled layout): each
// fragment is one contiguous 32-byte load per lane. K-steps are software
// pipelined in groups of 4: all 8 loads issued before the 4 WMMAs so the
// scheduler can overlap memory latency with the matrix pipe instead of a
// full counter drain per WMMA.
__device__ __forceinline__ v8f gemm_acc(const bf16* A, int lda, const bf16* Bnt,
                                        int Ksteps, int lane) {
  v8f acc = {0.f, 0.f, 0.f, 0.f, 0.f, 0.f, 0.f, 0.f};
  const bf16* bp = Bnt + lane * 16;
  for (int kc = 0; kc < Ksteps; kc += 4) {
    v16bf af[4], bf_[4];
#pragma unroll
    for (int u = 0; u < 4; ++u) {
      af[u]  = load_A(A, lda, (kc + u) * 32, lane);
      bf_[u] = *(const v16bf*)(bp + (size_t)(kc + u) * kTile);
    }
#pragma unroll
    for (int u = 0; u < 4; ++u) acc = wmma_bf16(af[u], bf_[u], acc);
  }
  return acc;
}
// C/D: lane -> N=lane&15 (+M half by lane>=16), VGPR j -> M = j (+8).
__device__ __forceinline__ void store_f32(float* C, int ldc, int nbase, int lane, v8f acc) {
  const int n  = nbase + (lane & 15);
  const int mo = (lane >> 4) * 8;
#pragma unroll
  for (int j = 0; j < 8; ++j) C[(mo + j) * ldc + n] = acc[j];
}
__device__ __forceinline__ void store_bf16(bf16* C, int ldc, int nbase, int lane, v8f acc) {
  const int n  = nbase + (lane & 15);
  const int mo = (lane >> 4) * 8;
#pragma unroll
  for (int j = 0; j < 8; ++j) C[(mo + j) * ldc + n] = (bf16)acc[j];
}

__device__ __forceinline__ float sigmoidf_(float v) { return 1.f / (1.f + expf(-v)); }

// ---------------- weight prep: f32 -> padded, fragment-swizzled bf16 ------
__device__ __forceinline__ void padcvt_sw(const float* s, int K, int N, bf16* d,
                                          int Kp, int Np, int tid, int stride) {
  const int KT = Kp / 32;
  const int tot = Kp * Np;
  for (int i = tid; i < tot; i += stride) {
    const int q    = i & 15;
    const int lane = (i >> 4) & 31;
    const int tile = i >> 9;
    const int kt   = tile % KT;
    const int nt   = tile / KT;
    const int n = nt * 16 + (lane & 15);
    const int k = kt * 32 + (lane >> 4) * 8 + (q & 7) + (q >> 3) * 16;
    d[i] = (k < K && n < N) ? (bf16)s[k * N + n] : (bf16)0.f;
  }
}

__global__ __launch_bounds__(kThreads)
void rim_prep_kernel(const float* __restrict__ Wk, const float* __restrict__ Wv,
                     const float* __restrict__ Wq, const float* __restrict__ Wih,
                     const float* __restrict__ Whh, const float* __restrict__ Wcq,
                     const float* __restrict__ Wck, const float* __restrict__ Wcv,
                     const float* __restrict__ Wco, bf16* __restrict__ ws) {
  const int tid = blockIdx.x * blockDim.x + threadIdx.x;
  const int nt  = gridDim.x * blockDim.x;
  padcvt_sw(Wk, 512, 64, ws + OFF_WK, 512, 64, tid, nt);
  padcvt_sw(Wv, 512, kVD, ws + OFF_WV, 512, kVP, tid, nt);
  padcvt_sw(Wq, kBS, 64, ws + OFF_WQ, kBSP, 64, tid, nt);
  for (int b = 0; b < kNB; ++b) {
    padcvt_sw(Wih + (size_t)b * kVD * (3 * kBS), kVD, 3 * kBS,
              ws + OFF_WIH + (size_t)b * kBSP * kG3P, kBSP, kG3P, tid, nt);
    padcvt_sw(Whh + (size_t)b * kBS * (3 * kBS), kBS, 3 * kBS,
              ws + OFF_WHH + (size_t)b * kBSP * kG3P, kBSP, kG3P, tid, nt);
  }
  padcvt_sw(Wcq, kBS, kCKV, ws + OFF_WCQ, kBSP, kCKV, tid, nt);
  padcvt_sw(Wck, kBS, kCKV, ws + OFF_WCK, kBSP, kCKV, tid, nt);
  padcvt_sw(Wcv, kBS, kCKV, ws + OFF_WCV, kBSP, kCKV, tid, nt);
  padcvt_sw(Wco, kCKV, kBS, ws + OFF_WCO, kCKV, kCOP, tid, nt);
}

// ---------------- persistent recurrent kernel ----------------
// 8 workgroups x 16 batch rows; hidden state flows through out[t-1] (L2).
__global__ __launch_bounds__(kThreads)
void rim_rnn_kernel(const float* __restrict__ x, const float* __restrict__ h0,
                    const float* __restrict__ bk, const float* __restrict__ bv,
                    const float* __restrict__ bq, const float* __restrict__ bih,
                    const float* __restrict__ bhh, const bf16* __restrict__ ws,
                    float* __restrict__ out, float* __restrict__ hlast) {
  // ---- overlaid LDS arena (97KB; CDNA5 WGP has 320KB LDS) ----
  __shared__ __align__(32) unsigned char lds[99072];
  bf16*  hb16  = (bf16*)lds;                       // [16][6*128] hidden (bf16, padded)
  float* p1s   = (float*)(lds + 24576);            // [16][6]
  float* masks = (float*)(lds + 24960);            // [16][6]
  unsigned char* scr = lds + 25344;                // 73728-byte phase scratch
  // phase A (input attention)
  bf16*  xs    = (bf16*)scr;                       // [16][512]
  float* kbuf  = (float*)(scr + 16384);            // [16][64]
  float* qbuf  = (float*)(scr + 20480);            // [16][6*64]
  float* vbuf  = (float*)(scr + 45056);            // [16][112]
  bf16*  att16 = (bf16*)scr;                       // [16][6*128] (reuses xs/k/q area)
  // phase B (GRU, one block at a time)
  float* gibuf = (float*)(scr + 24576);            // [16][304]
  float* ghbuf = (float*)(scr + 44032);            // [16][304]
  // phase C (communication attention)
  bf16* cq16 = (bf16*)scr;                         // [16][6*128]
  bf16* ck16 = (bf16*)(scr + 24576);
  bf16* cv16 = (bf16*)(scr + 49152);

  const int tid   = threadIdx.x;
  const int lane  = tid & 31;
  const int wid   = tid >> 5;
  const int bbase = blockIdx.x * kMB;

  const bf16* WkB  = ws + OFF_WK;
  const bf16* WvB  = ws + OFF_WV;
  const bf16* WqB  = ws + OFF_WQ;
  const bf16* WihB = ws + OFF_WIH;
  const bf16* WhhB = ws + OFF_WHH;
  const bf16* WcqB = ws + OFF_WCQ;
  const bf16* WckB = ws + OFF_WCK;
  const bf16* WcvB = ws + OFF_WCV;
  const bf16* WcoB = ws + OFF_WCO;

#pragma unroll 1
  for (int t = 0; t < kT; ++t) {
    __syncthreads();  // protect prior-iteration LDS readers before reuse
    // ---- stage x_t (b128 loads) and prefetch x_{t+1} ----
    const float* xrow = x + ((size_t)t * kB + bbase) * kNINP;
    const float4* x4  = (const float4*)xrow;
    for (int i = tid; i < kMB * kNINP / 4; i += kThreads) {
      const float4 f = x4[i];
      xs[4 * i + 0] = (bf16)f.x;
      xs[4 * i + 1] = (bf16)f.y;
      xs[4 * i + 2] = (bf16)f.z;
      xs[4 * i + 3] = (bf16)f.w;
    }
    if (t + 1 < kT) {  // hide next step's only cold HBM stream
      const float* xnext = x + ((size_t)(t + 1) * kB + bbase) * kNINP;
      for (int i = tid * 64; i < kMB * kNINP; i += kThreads * 64)
        __builtin_prefetch(xnext + i, 0, 3);
    }
    const float* hsrc = (t == 0) ? (h0 + (size_t)bbase * kH)
                                 : (out + ((size_t)(t - 1) * kB + bbase) * kH);
    for (int i = tid; i < kMB * kNB * kBSP; i += kThreads) {
      const int r = i / (kNB * kBSP), rr = i % (kNB * kBSP);
      const int blk = rr / kBSP, c = rr % kBSP;
      hb16[i] = (c < kBS) ? (bf16)hsrc[r * kH + blk * kBS + c] : (bf16)0.f;
    }
    __syncthreads();

    // ---- k = x@Wk (4 tiles), v = x@Wv (7 tiles), q = h@Wq per block (24) ----
    for (int j = wid; j < 4; j += 8)
      store_f32(kbuf, kKD, j * 16, lane,
                gemm_acc(xs, kNINP, WkB + (size_t)j * 16 * kTile, 16, lane));
    for (int j = wid; j < 7; j += 8)
      store_f32(vbuf, kVP, j * 16, lane,
                gemm_acc(xs, kNINP, WvB + (size_t)j * 16 * kTile, 16, lane));
    for (int j = wid; j < 24; j += 8) {
      const int blk = j >> 2, nt = j & 3;
      store_f32(qbuf + blk * kKD, kNB * kKD, nt * 16, lane,
                gemm_acc(hb16 + blk * kBSP, kNB * kBSP,
                         WqB + (size_t)nt * 4 * kTile, 4, lane));
    }
    __syncthreads();

    // ---- 2-way softmax over {null,input} per (b,block) ----
    if (tid < kMB * kNB) {
      const int b = tid & 15, blk = tid >> 4;
      float l0 = 0.f, l1 = 0.f;
      for (int d = 0; d < kKD; ++d) {
        const float qd = qbuf[b * (kNB * kKD) + blk * kKD + d] + bq[d];
        const float kn = bk[d];
        l0 += qd * kn;
        l1 += qd * (kbuf[b * kKD + d] + kn);
      }
      p1s[b * kNB + blk] = sigmoidf_((l1 - l0) * 0.125f);  // 1/sqrt(KD)=1/8
    }
    __syncthreads();
    // ---- top-4 of 6 by attention on real input ----
    if (tid < kMB) {
      float v[kNB], m[kNB];
      for (int n = 0; n < kNB; ++n) { v[n] = p1s[tid * kNB + n]; m[n] = 0.f; }
      for (int it = 0; it < kTOPK; ++it) {
        int best = 0; float bvv = v[0];
        for (int n = 1; n < kNB; ++n) if (v[n] > bvv) { bvv = v[n]; best = n; }
        m[best] = 1.f; v[best] = -1e30f;
      }
      for (int n = 0; n < kNB; ++n) masks[tid * kNB + n] = m[n];
    }
    __syncthreads();
    // ---- inp_att = bv + p1 * (x@Wv), padded bf16 (overlays xs/kbuf/qbuf) ----
    for (int i = tid; i < kMB * kNB * kBSP; i += kThreads) {
      const int b = i / (kNB * kBSP), rr = i % (kNB * kBSP);
      const int blk = rr / kBSP, c = rr % kBSP;
      float val = 0.f;
      if (c < kVD) val = bv[c] + p1s[b * kNB + blk] * vbuf[b * kVP + c];
      att16[i] = (bf16)val;
    }
    __syncthreads();

    // ---- block-diagonal GRU, one block at a time ----
    for (int blk = 0; blk < kNB; ++blk) {
      const bf16* Wi = WihB + (size_t)blk * kBSP * kG3P;
      const bf16* Wh = WhhB + (size_t)blk * kBSP * kG3P;
      for (int j = wid; j < 19; j += 8)
        store_f32(gibuf, kG3P, j * 16, lane,
                  gemm_acc(att16 + blk * kBSP, kNB * kBSP,
                           Wi + (size_t)j * 4 * kTile, 4, lane));
      for (int j = wid; j < 19; j += 8)
        store_f32(ghbuf, kG3P, j * 16, lane,
                  gemm_acc(hb16 + blk * kBSP, kNB * kBSP,
                           Wh + (size_t)j * 4 * kTile, 4, lane));
      __syncthreads();
      for (int i = tid; i < kMB * kBS; i += kThreads) {
        const int b = i / kBS, c = i % kBS;
        const float i_r = gibuf[b * kG3P + c]            + bih[blk * 3 * kBS + c];
        const float i_z = gibuf[b * kG3P + kBS + c]      + bih[blk * 3 * kBS + kBS + c];
        const float i_n = gibuf[b * kG3P + 2 * kBS + c]  + bih[blk * 3 * kBS + 2 * kBS + c];
        const float h_r = ghbuf[b * kG3P + c]            + bhh[blk * 3 * kBS + c];
        const float h_z = ghbuf[b * kG3P + kBS + c]      + bhh[blk * 3 * kBS + kBS + c];
        const float h_n = ghbuf[b * kG3P + 2 * kBS + c]  + bhh[blk * 3 * kBS + 2 * kBS + c];
        const float r = sigmoidf_(i_r + h_r);
        const float z = sigmoidf_(i_z + h_z);
        const float n = tanhf(i_n + r * h_n);
        const float hx = hsrc[b * kH + blk * kBS + c];
        const float hc = (1.f - z) * n + z * hx;
        const float m  = masks[b * kNB + blk];
        const float hn = m * hc + (1.f - m) * hx;
        out[((size_t)t * kB + bbase + b) * kH + blk * kBS + c] = hn;
        hb16[b * (kNB * kBSP) + blk * kBSP + c] = (bf16)hn;
      }
      __threadfence();  // make out[t] block visible for comm epilogue RMW
      __syncthreads();  // gi/gh reuse next block
    }

    // ---- communication attention: qc/kc/vc = hb @ Wc{q,k,v} ----
    for (int j = wid; j < 48; j += 8) {
      const int blk = j >> 3, nt = j & 7;
      store_bf16(cq16 + blk * kBSP, kNB * kBSP, nt * 16, lane,
                 gemm_acc(hb16 + blk * kBSP, kNB * kBSP,
                          WcqB + (size_t)nt * 4 * kTile, 4, lane));
    }
    for (int j = wid; j < 48; j += 8) {
      const int blk = j >> 3, nt = j & 7;
      store_bf16(ck16 + blk * kBSP, kNB * kBSP, nt * 16, lane,
                 gemm_acc(hb16 + blk * kBSP, kNB * kBSP,
                          WckB + (size_t)nt * 4 * kTile, 4, lane));
    }
    for (int j = wid; j < 48; j += 8) {
      const int blk = j >> 3, nt = j & 7;
      store_bf16(cv16 + blk * kBSP, kNB * kBSP, nt * 16, lane,
                 gemm_acc(hb16 + blk * kBSP, kNB * kBSP,
                          WcvB + (size_t)nt * 4 * kTile, 4, lane));
    }
    __syncthreads();
    // ---- per-(b,block) 6-way softmax over heads; comm vec written over cq16 ----
    if (tid < kMB * kNB) {
      const int b = tid & 15, nblk = tid >> 4;
      float pc[kCH][kNB];
      for (int h = 0; h < kCH; ++h) {
        float s[kNB], mx = -1e30f;
        for (int m = 0; m < kNB; ++m) {
          float d = 0.f;
          for (int c = 0; c < kCK; ++c)
            d += (float)cq16[b * (kNB * kBSP) + nblk * kBSP + h * kCK + c] *
                 (float)ck16[b * (kNB * kBSP) + m * kBSP + h * kCK + c];
          s[m] = d * 0.17677669529663687f;  // 1/sqrt(CK)
          if (s[m] > mx) mx = s[m];
        }
        float den = 0.f;
        for (int m = 0; m < kNB; ++m) { s[m] = expf(s[m] - mx); den += s[m]; }
        const float inv = 1.f / den;
        for (int m = 0; m < kNB; ++m) pc[h][m] = s[m] * inv;
      }
      for (int c = 0; c < kCKV; ++c) {
        const int h = c >> 5;
        float acc = 0.f;
        for (int m = 0; m < kNB; ++m)
          acc += pc[h][m] * (float)cv16[b * (kNB * kBSP) + m * kBSP + c];
        cq16[b * (kNB * kBSP) + nblk * kBSP + c] = (bf16)acc;  // own row: safe
      }
    }
    __syncthreads();
    // ---- comm @ Wco, masked residual add into out[t] ----
    for (int j = wid; j < 42; j += 8) {
      const int blk = j / 7, nt = j % 7;
      v8f acc = gemm_acc(cq16 + blk * kBSP, kNB * kBSP,
                         WcoB + (size_t)nt * 4 * kTile, 4, lane);
      const int n  = nt * 16 + (lane & 15);
      const int mo = (lane >> 4) * 8;
      if (n < kBS) {
#pragma unroll
        for (int jj = 0; jj < 8; ++jj) {
          const int r = mo + jj;
          const size_t o = ((size_t)t * kB + bbase + r) * kH + blk * kBS + n;
          out[o] += masks[r * kNB + blk] * acc[jj];
        }
      }
    }
    __threadfence();  // out[t] complete before next step reads it as h_prev

    if (t == kT - 1) {
      __syncthreads();
      for (int i = tid; i < kMB * kH; i += kThreads) {
        const int r = i / kH, c = i % kH;
        hlast[(size_t)(bbase + r) * kH + c] =
            out[((size_t)(kT - 1) * kB + bbase + r) * kH + c];
      }
    }
  }
}

extern "C" void kernel_launch(void* const* d_in, const int* in_sizes, int n_in,
                              void* d_out, int out_size, void* d_ws, size_t ws_size,
                              hipStream_t stream) {
  const float* x   = (const float*)d_in[0];
  const float* h0  = (const float*)d_in[1];
  const float* Wk  = (const float*)d_in[2];
  const float* bk  = (const float*)d_in[3];
  const float* Wv  = (const float*)d_in[4];
  const float* bv  = (const float*)d_in[5];
  const float* Wq  = (const float*)d_in[6];
  const float* bq  = (const float*)d_in[7];
  const float* Wih = (const float*)d_in[8];
  const float* bih = (const float*)d_in[9];
  const float* Whh = (const float*)d_in[10];
  const float* bhh = (const float*)d_in[11];
  const float* Wcq = (const float*)d_in[12];
  const float* Wck = (const float*)d_in[13];
  const float* Wcv = (const float*)d_in[14];
  const float* Wco = (const float*)d_in[15];
  float* out   = (float*)d_out;
  float* hlast = out + (size_t)kT * kB * kH;
  bf16*  ws    = (bf16*)d_ws;  // needs ~1.26 MB

  rim_prep_kernel<<<64, kThreads, 0, stream>>>(Wk, Wv, Wq, Wih, Whh, Wcq, Wck, Wcv, Wco, ws);
  rim_rnn_kernel<<<kB / kMB, kThreads, 0, stream>>>(x, h0, bk, bv, bq, bih, bhh, ws, out, hlast);
}
